// SampledSoftmaxLoss_927712936472
// MI455X (gfx1250) — matrix operand order repaired and checked by reading, not compiled, not run on hardware
//
#include <hip/hip_runtime.h>
#include <hip/hip_bf16.h>
#include <stdint.h>

typedef __attribute__((ext_vector_type(16))) __bf16 v16bf;
typedef __attribute__((ext_vector_type(8)))  float  v8f;

// ---------------------------------------------------------------------------
// fp32 -> bf16 (round to nearest even)
// ---------------------------------------------------------------------------
static __device__ __forceinline__ unsigned short f32_to_bf16(float f) {
    uint32_t u = __float_as_uint(f);
    uint32_t r = u + 0x7FFFu + ((u >> 16) & 1u);
    return (unsigned short)(r >> 16);
}

__global__ void convert_bf16_kernel(const float* __restrict__ a,
                                    const float* __restrict__ b,
                                    unsigned short* __restrict__ abf,
                                    unsigned short* __restrict__ bbf,
                                    int total) {
    int i = blockIdx.x * blockDim.x + threadIdx.x;
    if (i < total) {
        abf[i] = f32_to_bf16(a[i]);
        bbf[i] = f32_to_bf16(b[i]);
    }
}

// ---------------------------------------------------------------------------
// Per-row stats: pos_sim, row_has_su, scale = mask ? (1-q)/max(n_miss,1) : 0
// Also packs per-COLUMN metadata {tid, uid, notmask, bits(1/q)} into one int4
// so the hot loop does a single B128 load per column tile.
// ---------------------------------------------------------------------------
__global__ void row_stats_kernel(const float* __restrict__ in_emb,
                                 const float* __restrict__ tg_emb,
                                 const float* __restrict__ q,
                                 const int*   __restrict__ tids,
                                 const int*   __restrict__ uids,
                                 const unsigned char* __restrict__ mask,
                                 float* __restrict__ pos_sim,
                                 float* __restrict__ scale,
                                 int*   __restrict__ has_su_out,
                                 int4*  __restrict__ colmeta,
                                 int N, int D) {
    __shared__ int s_tid[256];
    __shared__ int s_uid[256];
    __shared__ int s_nm[256];

    const int i = blockIdx.x * 256 + threadIdx.x;   // one row per thread

    float p = 0.0f;
    for (int d = 0; d < D; ++d)
        p = fmaf(in_emb[(size_t)i * D + d], tg_emb[(size_t)i * D + d], p);

    const int mytid = tids[i];
    const int myuid = uids[i];
    const int mynm  = mask[i] ? 0 : 1;
    int cnt_all = 0, cnt_su = 0, has = 0;

    for (int base = 0; base < N; base += 256) {
        s_tid[threadIdx.x] = tids[base + threadIdx.x];
        s_uid[threadIdx.x] = uids[base + threadIdx.x];
        s_nm[threadIdx.x]  = mask[base + threadIdx.x] ? 0 : 1;
        __syncthreads();
        for (int kk = 0; kk < 256; ++kk) {
            int notm = s_nm[kk];
            int dif  = (mytid != s_tid[kk]) ? 1 : 0;
            int ueq  = (myuid == s_uid[kk]) ? 1 : 0;
            has     |= (ueq & notm);
            cnt_all += (dif & notm);
            cnt_su  += (dif & notm & ueq);
        }
        __syncthreads();
    }

    const int   mi     = mask[i] ? 1 : 0;
    const int   n      = has ? cnt_su : cnt_all;
    const float safe_n = fmaxf((float)n, 1.0f);

    pos_sim[i]    = p;
    scale[i]      = mi ? (1.0f - q[i]) / safe_n : 0.0f;
    has_su_out[i] = has;

    int4 cm;
    cm.x = mytid;
    cm.y = myuid;
    cm.z = mynm;                               // !mask[i] as int
    cm.w = __float_as_int(1.0f / q[i]);        // reciprocal hoisted here
    colmeta[i] = cm;
}

// ---------------------------------------------------------------------------
// Fused GEMM (bf16 WMMA) + masked exp epilogue. One wave owns one 16-row
// tile and sweeps all column tiles 2 at a time (software pipelined, separate
// accumulators) so independent WMMA/loads fill the WMMA->VALU hazard slots
// and hide v_exp_f32 latency. Accumulates  sum_k miss*exp(S)/q_k.
// ---------------------------------------------------------------------------
__global__ void __launch_bounds__(256)
neg_softmax_kernel(const unsigned short* __restrict__ Abf,
                   const unsigned short* __restrict__ Bbf,
                   const int4*  __restrict__ colmeta,
                   const int*   __restrict__ tids,
                   const int*   __restrict__ uids,
                   const unsigned char* __restrict__ mask,
                   const float* __restrict__ pos_sim,
                   const float* __restrict__ scale,
                   const int*   __restrict__ has_su,
                   float* __restrict__ per_row,
                   int N) {
    const int lane = threadIdx.x & 31;
    const int wave = threadIdx.x >> 5;
    const int h    = lane >> 4;     // which half of the wave
    const int n    = lane & 15;     // tile-local column (B) / row (A)
    const int rowBase = (blockIdx.x * 8 + wave) * 16;

    union Frag { uint4 u[2]; v16bf v; };

    // A fragments for row rowBase+n (row-major, K-contiguous bf16):
    //   lo (K=0..31):  VGPR0-3 <- K=8h..8h+7,  VGPR4-7 <- K=16+8h..23+8h
    //   hi (K=32..63): same pattern +32
    const unsigned short* arow = Abf + (size_t)(rowBase + n) * 64;
    Frag aLo, aHi;
    aLo.u[0] = *(const uint4*)(arow +      8 * h);
    aLo.u[1] = *(const uint4*)(arow + 16 + 8 * h);
    aHi.u[0] = *(const uint4*)(arow + 32 + 8 * h);
    aHi.u[1] = *(const uint4*)(arow + 48 + 8 * h);

    // Per-row metadata for the 8 C rows this lane touches (M = j + 8h)
    int tidr[8], uidr[8], hasr[8], mr[8];
    float sclr[8];
#pragma unroll
    for (int j = 0; j < 8; ++j) {
        int r   = rowBase + j + 8 * h;
        tidr[j] = tids[r];
        uidr[j] = uids[r];
        hasr[j] = has_su[r];
        sclr[j] = scale[r];
        mr[j]   = mask[r] ? 1 : 0;
    }

    float acc0[8], acc1[8];
#pragma unroll
    for (int j = 0; j < 8; ++j) { acc0[j] = 0.0f; acc1[j] = 0.0f; }

    const int numColTiles = N >> 4;          // 512 (even)
    for (int ct = 0; ct < numColTiles; ct += 2) {
        const int col0 = (ct << 4) + n;
        const int col1 = col0 + 16;
        const unsigned short* brow0 = Bbf + (size_t)col0 * 64;
        const unsigned short* brow1 = Bbf + (size_t)col1 * 64;

        if (ct + 2 < numColTiles)
            __builtin_prefetch(Bbf + (size_t)(col0 + 32) * 64, 0, 0);

        // B fragments (lo: K=0..31, hi: K=32..63) for both column tiles:
        //   lo: VGPR0-3 <- K=8h..8h+7, VGPR4-7 <- K=16+8h..23+8h
        Frag b0Lo, b0Hi, b1Lo, b1Hi;
        b0Lo.u[0] = *(const uint4*)(brow0 + 8 * h);
        b0Lo.u[1] = *(const uint4*)(brow0 + 16 + 8 * h);
        b0Hi.u[0] = *(const uint4*)(brow0 + 32 + 8 * h);
        b0Hi.u[1] = *(const uint4*)(brow0 + 48 + 8 * h);
        b1Lo.u[0] = *(const uint4*)(brow1 + 8 * h);
        b1Lo.u[1] = *(const uint4*)(brow1 + 16 + 8 * h);
        b1Hi.u[0] = *(const uint4*)(brow1 + 32 + 8 * h);
        b1Hi.u[1] = *(const uint4*)(brow1 + 48 + 8 * h);

        // Single B128 load per tile for column predicates + 1/q
        const int4 cm0 = colmeta[col0];
        const int4 cm1 = colmeta[col1];

        v8f c0 = {0.0f, 0.0f, 0.0f, 0.0f, 0.0f, 0.0f, 0.0f, 0.0f};
        v8f c1 = {0.0f, 0.0f, 0.0f, 0.0f, 0.0f, 0.0f, 0.0f, 0.0f};
        c0 = __builtin_amdgcn_wmma_f32_16x16x32_bf16(false, aLo.v, false, b0Lo.v,
                                                     (short)0, c0, false, false);
        c1 = __builtin_amdgcn_wmma_f32_16x16x32_bf16(false, aLo.v, false, b1Lo.v,
                                                     (short)0, c1, false, false);
        c0 = __builtin_amdgcn_wmma_f32_16x16x32_bf16(false, aHi.v, false, b0Hi.v,
                                                     (short)0, c0, false, false);
        c1 = __builtin_amdgcn_wmma_f32_16x16x32_bf16(false, aHi.v, false, b1Hi.v,
                                                     (short)0, c1, false, false);

        const float rq0 = __int_as_float(cm0.w);
        const float rq1 = __int_as_float(cm1.w);

        // C layout: element j of c is S[rowBase + j + 8h][col]
#pragma unroll
        for (int j = 0; j < 8; ++j) {
            int miss0 = mr[j] & cm0.z & ((tidr[j] != cm0.x) ? 1 : 0) &
                        (((hasr[j] == 0) ? 1 : 0) | ((uidr[j] == cm0.y) ? 1 : 0));
            int miss1 = mr[j] & cm1.z & ((tidr[j] != cm1.x) ? 1 : 0) &
                        (((hasr[j] == 0) ? 1 : 0) | ((uidr[j] == cm1.y) ? 1 : 0));
            float e0 = __expf(c0[j]) * rq0;
            float e1 = __expf(c1[j]) * rq1;
            acc0[j] += miss0 ? e0 : 0.0f;
            acc1[j] += miss1 ? e1 : 0.0f;
        }
    }

    // Merge the two pipelined accumulators, then reduce each row sum across
    // the 16 lanes of its half (xor masks <16 stay within the half).
#pragma unroll
    for (int j = 0; j < 8; ++j) {
        float v = acc0[j] + acc1[j];
        v += __shfl_xor(v, 1, 32);
        v += __shfl_xor(v, 2, 32);
        v += __shfl_xor(v, 4, 32);
        v += __shfl_xor(v, 8, 32);
        acc0[j] = v;
    }

    if (n == 0) {  // lanes 0 and 16 hold rows j+0..7 / j+8..15
#pragma unroll
        for (int j = 0; j < 8; ++j) {
            int   r   = rowBase + j + 8 * h;
            float neg = acc0[j] * sclr[j];         // (1-q_r)/safe_n applied here
            float p   = pos_sim[r];
            per_row[r] = mr[j] ? (logf(__expf(p) + neg) - p) : 0.0f;
        }
    }
}

// ---------------------------------------------------------------------------
// Deterministic final reduction: loss = sum(per_row) / sum(mask)
// ---------------------------------------------------------------------------
__global__ void final_reduce_kernel(const float* __restrict__ per_row,
                                    const unsigned char* __restrict__ mask,
                                    float* __restrict__ out, int N) {
    __shared__ float ss[256];
    __shared__ float sc[256];
    float s = 0.0f, c = 0.0f;
    for (int i = threadIdx.x; i < N; i += 256) {
        s += per_row[i];
        c += mask[i] ? 1.0f : 0.0f;
    }
    ss[threadIdx.x] = s;
    sc[threadIdx.x] = c;
    __syncthreads();
    for (int o = 128; o > 0; o >>= 1) {
        if (threadIdx.x < o) {
            ss[threadIdx.x] += ss[threadIdx.x + o];
            sc[threadIdx.x] += sc[threadIdx.x + o];
        }
        __syncthreads();
    }
    if (threadIdx.x == 0) out[0] = ss[0] / sc[0];
}

// ---------------------------------------------------------------------------
extern "C" void kernel_launch(void* const* d_in, const int* in_sizes, int n_in,
                              void* d_out, int out_size, void* d_ws, size_t ws_size,
                              hipStream_t stream) {
    const float*         in_emb = (const float*)d_in[0];
    const float*         tg_emb = (const float*)d_in[1];
    const float*         q      = (const float*)d_in[2];
    const int*           tids   = (const int*)d_in[3];
    const int*           uids   = (const int*)d_in[4];
    const unsigned char* mask   = (const unsigned char*)d_in[5];
    float*               out    = (float*)d_out;

    const int N = in_sizes[2];          // 8192
    const int D = in_sizes[0] / N;      // 64 (tiling assumes 64)

    // Workspace layout
    unsigned short* Abf = (unsigned short*)d_ws;              // N*D bf16
    unsigned short* Bbf = Abf + (size_t)N * D;                // N*D bf16
    float* pos = (float*)(Bbf + (size_t)N * D);               // N f32
    float* scl = pos + N;                                     // N f32
    int*   has = (int*)(scl + N);                             // N i32
    float* per = (float*)(has + N);                           // N f32
    int4*  cmeta = (int4*)(per + N);                          // N int4 (16B aligned)

    const int total = N * D;
    convert_bf16_kernel<<<(total + 255) / 256, 256, 0, stream>>>(
        in_emb, tg_emb, Abf, Bbf, total);

    row_stats_kernel<<<N / 256, 256, 0, stream>>>(
        in_emb, tg_emb, q, tids, uids, mask, pos, scl, has, cmeta, N, D);

    // 512 row tiles of 16 rows, 8 waves (tiles) per 256-thread block
    neg_softmax_kernel<<<(N / 16) / 8, 256, 0, stream>>>(
        Abf, Bbf, cmeta, tids, uids, mask, pos, scl, has, per, N);

    final_reduce_kernel<<<1, 256, 0, stream>>>(per, mask, out, N);
}
